// FLASH_19550691131611
// MI455X (gfx1250) — compile-verified, matrix operand order checked by reading, not run on hardware
//
#include <hip/hip_runtime.h>
#include <cstdint>

#define USE_TDM 1

// ---------------------------------------------------------------------------
// Types for CDNA5 WMMA / TDM
// ---------------------------------------------------------------------------
typedef __attribute__((ext_vector_type(16))) _Float16 v16h;
typedef __attribute__((ext_vector_type(8)))  float    v8f;
typedef __attribute__((ext_vector_type(4)))  unsigned int v4u;
typedef __attribute__((ext_vector_type(8)))  unsigned int v8u;
using h16 = _Float16;

struct __attribute__((aligned(16))) U4 { unsigned int a, b, c, d; };
union U4H { U4 q; h16 h[8]; unsigned int u[4]; };
union Frag { v16h v; unsigned int u[8]; };

// Tile config: 256 threads = 8 wave32, block tile 128x128, K-step 32.
// Waves arranged 4 (M) x 2 (N); each wave owns a 32x64 sub-tile = 2x4 accums.
constexpr int BM = 128, BN = 128, BKt = 32, LDK = 34; // LDK = 32 + 2 pad halfs

// ---------------------------------------------------------------------------
// Tensor Data Mover: DMA a rows x 32(f16) tile from global into LDS with
// hardware padding of 1 DWORD every 16 DWORDs -> LDS row stride 34 halfs.
// D# groups 0/1 only (2D tensor); issued per-wave for a 16-row slice.
// ---------------------------------------------------------------------------
__device__ __forceinline__ unsigned lds_off_u32(const void* p) {
    // generic LDS pointer: addr[31:0] is the LDS byte offset
    return (unsigned)(unsigned long long)p;
}

__device__ __forceinline__ void tdm_load_tile(const h16* g, unsigned lds_byte_off,
                                              int ld_elems, int rows) {
    unsigned long long ga = (unsigned long long)g;
    v4u g0;
    g0[0] = 1u;                                   // count=1 (valid descriptor)
    g0[1] = lds_byte_off;                         // lds_addr
    g0[2] = (unsigned)ga;                         // global_addr[31:0]
    g0[3] = (unsigned)((ga >> 32) & 0x01FFFFFFull) | 0x80000000u; // addr[56:32] | type=2
    v8u g1;
    g1[0] = (1u << 16)      // data_size = 2 bytes
          | (1u << 20)      // pad_enable
          | (3u << 22);     // pad_interval: every 16 DWORDs (=64B = one 32-elem row)
                            // pad_amount field 0 => 1 DWORD pad  -> stride 34 halfs
    g1[1] = 0u;                                   // tensor_dim0 = 1<<30 (low16 = 0)
    g1[2] = 0x4000u;                              // tensor_dim0 hi | tensor_dim1 low (1<<30)
    g1[3] = 0x4000u | ((unsigned)BKt << 16);      // tensor_dim1 hi | tile_dim0 = 32
    g1[4] = (unsigned)rows;                       // tile_dim1 = rows, tile_dim2 = 0
    g1[5] = (unsigned)ld_elems;                   // tensor_dim0_stride (elements)
    g1[6] = 0u;
    g1[7] = 0u;
    asm volatile("tensor_load_to_lds %0, %1" :: "s"(g0), "s"(g1) : "memory");
}

// ---------------------------------------------------------------------------
// Generic WMMA GEMM core.  AT: A stored K x M (transpose-stage).  BT: B stored
// N x K (compute A * B^T).  Double-buffered LDS, software-pipelined staging.
// ---------------------------------------------------------------------------
template <bool AT, bool BT, class Epi>
__global__ __launch_bounds__(256) void gemm_wmma(
    const h16* __restrict__ Abase, const h16* __restrict__ Bbase,
    int M, int N, int K, int lda, int ldb,
    long long strideA, long long strideB, Epi epi)
{
    __shared__ h16 aS[2][BM * LDK];
    __shared__ h16 bS[2][BN * LDK];

    const int z = blockIdx.z;
    const h16* A  = Abase + (long long)z * strideA;
    const h16* Bp = Bbase + (long long)z * strideB;

    const int tileM = blockIdx.y * BM;
    const int tileN = blockIdx.x * BN;
    const int tid   = threadIdx.x;
    const int lane  = tid & 31;
    const int wave  = tid >> 5;
    const int wm    = wave & 3;     // 0..3  (M direction)
    const int wn    = wave >> 2;    // 0..1  (N direction)
    const int lrow  = lane & 15;
    const int khalf = lane >> 4;    // 0 or 1

    constexpr bool TDM_A = !AT && (USE_TDM != 0);
    constexpr bool TDM_B =  BT && (USE_TDM != 0);
    constexpr bool ANY_TDM = TDM_A || TDM_B;

    // ---- phase: issue global reads (regs) or TDM (direct to LDS[buf]) ----
    auto loadA = [&](int buf, int k0, U4H ra[2]) {
        if constexpr (!AT) {
            if constexpr (TDM_A) {
                tdm_load_tile(A + (long long)(tileM + wave * 16) * lda + k0,
                              lds_off_u32(&aS[buf][wave * 16 * LDK]), lda, 16);
            } else {
#pragma unroll
                for (int it = 0; it < 2; ++it) {
                    int idx = it * 256 + tid;
                    int r = idx >> 2, c8 = (idx & 3) << 3;
                    ra[it].q = *reinterpret_cast<const U4*>(
                        A + (long long)(tileM + r) * lda + k0 + c8);
                }
            }
        } else { // A stored K x M: read 8 contiguous along M, scatter later
#pragma unroll
            for (int it = 0; it < 2; ++it) {
                int idx = it * 256 + tid;
                int kk = idx >> 4, m8 = (idx & 15) << 3;
                ra[it].q = *reinterpret_cast<const U4*>(
                    A + (long long)(k0 + kk) * lda + tileM + m8);
            }
        }
    };
    auto loadB = [&](int buf, int k0, U4H rb[2]) {
        if constexpr (BT) {
            if constexpr (TDM_B) {
                tdm_load_tile(Bp + (long long)(tileN + wave * 16) * ldb + k0,
                              lds_off_u32(&bS[buf][wave * 16 * LDK]), ldb, 16);
            } else {
#pragma unroll
                for (int it = 0; it < 2; ++it) {
                    int idx = it * 256 + tid;
                    int r = idx >> 2, c8 = (idx & 3) << 3;
                    rb[it].q = *reinterpret_cast<const U4*>(
                        Bp + (long long)(tileN + r) * ldb + k0 + c8);
                }
            }
        } else { // B stored K x N: read 8 contiguous along N, scatter later
#pragma unroll
            for (int it = 0; it < 2; ++it) {
                int idx = it * 256 + tid;
                int kk = idx >> 4, n8 = (idx & 15) << 3;
                rb[it].q = *reinterpret_cast<const U4*>(
                    Bp + (long long)(k0 + kk) * ldb + tileN + n8);
            }
            if (k0 + BKt < K) // prefetch next K-tile -> global_prefetch_b8
                __builtin_prefetch(Bp + (long long)(k0 + BKt + (tid >> 4)) * ldb
                                       + tileN + ((tid & 15) << 3), 0, 0);
        }
    };
    // ---- phase: commit register-staged data into LDS[buf] ----
    auto commitA = [&](int buf, U4H ra[2]) {
        if constexpr (!AT) {
            if constexpr (!TDM_A) {
#pragma unroll
                for (int it = 0; it < 2; ++it) {
                    int idx = it * 256 + tid;
                    int r = idx >> 2, c8 = (idx & 3) << 3;
                    unsigned int* dst =
                        reinterpret_cast<unsigned int*>(&aS[buf][r * LDK + c8]);
                    dst[0] = ra[it].u[0]; dst[1] = ra[it].u[1];
                    dst[2] = ra[it].u[2]; dst[3] = ra[it].u[3];
                }
            }
        } else {
#pragma unroll
            for (int it = 0; it < 2; ++it) {
                int idx = it * 256 + tid;
                int kk = idx >> 4, m8 = (idx & 15) << 3;
#pragma unroll
                for (int j = 0; j < 8; ++j) aS[buf][(m8 + j) * LDK + kk] = ra[it].h[j];
            }
        }
    };
    auto commitB = [&](int buf, U4H rb[2]) {
        if constexpr (BT) {
            if constexpr (!TDM_B) {
#pragma unroll
                for (int it = 0; it < 2; ++it) {
                    int idx = it * 256 + tid;
                    int r = idx >> 2, c8 = (idx & 3) << 3;
                    unsigned int* dst =
                        reinterpret_cast<unsigned int*>(&bS[buf][r * LDK + c8]);
                    dst[0] = rb[it].u[0]; dst[1] = rb[it].u[1];
                    dst[2] = rb[it].u[2]; dst[3] = rb[it].u[3];
                }
            }
        } else {
#pragma unroll
            for (int it = 0; it < 2; ++it) {
                int idx = it * 256 + tid;
                int kk = idx >> 4, n8 = (idx & 15) << 3;
#pragma unroll
                for (int j = 0; j < 8; ++j) bS[buf][(n8 + j) * LDK + kk] = rb[it].h[j];
            }
        }
    };

    v8f c[2][4];
#pragma unroll
    for (int i = 0; i < 2; ++i)
#pragma unroll
        for (int j = 0; j < 4; ++j)
#pragma unroll
            for (int e = 0; e < 8; ++e) c[i][j][e] = 0.0f;

    U4H ra[2], rb[2];

    // prologue: stage tile 0 into buffer 0
    loadA(0, 0, ra); loadB(0, 0, rb);
    commitA(0, ra);  commitB(0, rb);
#if USE_TDM
    if constexpr (ANY_TDM) __builtin_amdgcn_s_wait_tensorcnt(0);
#endif
    __syncthreads();

    int cur = 0;
    for (int k0 = 0; k0 < K; k0 += BKt) {
        const bool has_next = (k0 + BKt) < K;
        if (has_next) { loadA(cur ^ 1, k0 + BKt, ra); loadB(cur ^ 1, k0 + BKt, rb); }

        // ---- fragments per ISA 7.12.2 16-bit layout, then 8 WMMAs ----
        Frag af[2], bf[4];
#pragma unroll
        for (int mi = 0; mi < 2; ++mi) {
            int r = wm * 32 + mi * 16 + lrow;
            const h16* base = &aS[cur][r * LDK];
#pragma unroll
            for (int i = 0; i < 8; ++i) {
                int kk = ((i & 3) << 1) | (khalf << 3) | ((i >> 2) << 4);
                af[mi].u[i] = *reinterpret_cast<const unsigned int*>(base + kk);
            }
        }
#pragma unroll
        for (int ni = 0; ni < 4; ++ni) {
            int col = wn * 64 + ni * 16 + lrow;
            const h16* base = &bS[cur][col * LDK];
#pragma unroll
            for (int i = 0; i < 8; ++i) {
                int kk = ((i & 3) << 1) | (khalf << 3) | ((i >> 2) << 4);
                bf[ni].u[i] = *reinterpret_cast<const unsigned int*>(base + kk);
            }
        }
#pragma unroll
        for (int mi = 0; mi < 2; ++mi)
#pragma unroll
            for (int ni = 0; ni < 4; ++ni)
                c[mi][ni] = __builtin_amdgcn_wmma_f32_16x16x32_f16(
                    false, af[mi].v, false, bf[ni].v, (short)0, c[mi][ni], false, false);

        if (has_next) { commitA(cur ^ 1, ra); commitB(cur ^ 1, rb); }
#if USE_TDM
        if constexpr (ANY_TDM) __builtin_amdgcn_s_wait_tensorcnt(0);
#endif
        __syncthreads();
        cur ^= 1;
    }

    // ---- epilogue: C dword j -> row j (lanes 0-15) / j+8 (lanes 16-31) ----
#pragma unroll
    for (int mi = 0; mi < 2; ++mi)
#pragma unroll
        for (int ni = 0; ni < 4; ++ni) {
            int gn = tileN + wn * 64 + ni * 16 + lrow;
#pragma unroll
            for (int j = 0; j < 8; ++j) {
                int gm = tileM + wm * 32 + mi * 16 + khalf * 8 + j;
                if (gm < M && gn < N) epi(z, gm, gn, c[mi][ni][j]);
            }
        }
}

// ---------------------------------------------------------------------------
// Fused epilogues
// ---------------------------------------------------------------------------
struct EpiSiluH16 {  // hidden = silu(x@W_h + b_h), stored f16
    h16* dst; const float* bias; int ldo;
    __device__ void operator()(int, int m, int n, float v) const {
        v += bias[n];
        v = v / (1.0f + __expf(-v));
        dst[(long long)m * ldo + n] = (h16)v;
    }
};
struct EpiSiluF32 {  // qk = silu(x@W_qk + b_qk), stored f32
    float* dst; const float* bias; int ldo;
    __device__ void operator()(int, int m, int n, float v) const {
        v += bias[n];
        dst[(long long)m * ldo + n] = v / (1.0f + __expf(-v));
    }
};
struct EpiAttn {     // attn = relu(sim/G + t5_bias)^2, stored f16
    h16* dst; const float* tab;
    __device__ void operator()(int z, int m, int n, float v) const {
        float s = v * (1.0f / 256.0f) + tab[m - n + 255];
        s = fmaxf(s, 0.0f);
        dst[(long long)z * 65536 + m * 256 + n] = (h16)(s * s);
    }
};
struct EpiScaleH16 { // scaled f16 store, batched row offset
    h16* dst; float scale; int ldo; int zrs;
    __device__ void operator()(int z, int m, int n, float v) const {
        dst[((long long)z * zrs + m) * ldo + n] = (h16)(v * scale);
    }
};
struct EpiFinal {    // out = acc + b_o + x (residual), f32
    float* dst; const float* bias; const float* xres; int ldo;
    __device__ void operator()(int, int m, int n, float v) const {
        long long idx = (long long)m * ldo + n;
        dst[idx] = v + bias[n] + xres[idx];
    }
};

// ---------------------------------------------------------------------------
// Small elementwise / reduction kernels
// ---------------------------------------------------------------------------
__global__ __launch_bounds__(256) void k_cast_f32_h16(const float* __restrict__ s,
                                                      h16* __restrict__ d, long long n) {
    long long i = (long long)blockIdx.x * 256 + threadIdx.x;
    if (i < n) d[i] = (h16)s[i];
}

__global__ __launch_bounds__(256) void k_layernorm(const float* __restrict__ x,
                                                   const float* __restrict__ g,
                                                   const float* __restrict__ b,
                                                   h16* __restrict__ out, int D) {
    int row = blockIdx.x;
    const float* xr = x + (long long)row * D;
    float s = 0.f, s2 = 0.f;
    for (int i = threadIdx.x; i < D; i += 256) { float v = xr[i]; s += v; s2 += v * v; }
    __shared__ float rs[256], rq[256];
    rs[threadIdx.x] = s; rq[threadIdx.x] = s2;
    __syncthreads();
    for (int off = 128; off > 0; off >>= 1) {
        if (threadIdx.x < off) { rs[threadIdx.x] += rs[threadIdx.x + off];
                                 rq[threadIdx.x] += rq[threadIdx.x + off]; }
        __syncthreads();
    }
    float mu  = rs[0] / D;
    float var = rq[0] / D - mu * mu;
    float inv = rsqrtf(var + 1e-5f);
    for (int i = threadIdx.x; i < D; i += 256)
        out[(long long)row * D + i] = (h16)((xr[i] - mu) * inv * g[i] + b[i]);
}

__global__ void k_relbias(const float* __restrict__ rel_emb, float* __restrict__ table) {
    int i = threadIdx.x;            // one block of 512; entries for rel = i-255
    if (i >= 511) return;
    int rel = i - 255;              // rel = (query i) - (key j)
    int ret = (rel < 0) ? 16 : 0;   // NUM_BUCKETS/2 = 16
    int a = rel < 0 ? -rel : rel;
    int bucket;
    if (a < 8) bucket = ret + a;    // max_exact = 8
    else {
        int vl = 8 + (int)(logf((float)a / 8.0f) / logf(16.0f) * 8.0f);
        if (vl > 15) vl = 15;
        bucket = ret + vl;
    }
    table[i] = rel_emb[bucket] * sqrtf(128.0f); // SCALE = sqrt(QK)
}

__global__ __launch_bounds__(256) void k_heads(const float* __restrict__ qk,
                                               const float* __restrict__ os_g,
                                               const float* __restrict__ os_b,
                                               h16* __restrict__ qq, h16* __restrict__ lq,
                                               h16* __restrict__ qkh, h16* __restrict__ lk) {
    long long i = (long long)blockIdx.x * 256 + threadIdx.x; // 16384*128 elems
    int d = (int)(i & 127);
    float v = qk[i];
    qq[i]  = (h16)(v * os_g[0 * 128 + d] + os_b[0 * 128 + d]);
    lq[i]  = (h16)(v * os_g[1 * 128 + d] + os_b[1 * 128 + d]);
    qkh[i] = (h16)(v * os_g[2 * 128 + d] + os_b[2 * 128 + d]);
    lk[i]  = (h16)(v * os_g[3 * 128 + d] + os_b[3 * 128 + d]);
}

__global__ __launch_bounds__(256) void k_combine(const h16* __restrict__ hidden,
                                                 const h16* __restrict__ l,
                                                 h16* __restrict__ q /* in/out */) {
    long long i = (long long)blockIdx.x * 256 + threadIdx.x; // 16384*2048
    long long m = i >> 11;
    int e = (int)(i & 2047);
    float gate = (float)hidden[m * 4096 + 2048 + e];
    q[i] = (h16)(gate * ((float)q[i] + (float)l[i]));
}

// ---------------------------------------------------------------------------
// Host orchestration
// ---------------------------------------------------------------------------
extern "C" void kernel_launch(void* const* d_in, const int* in_sizes, int n_in,
                              void* d_out, int out_size, void* d_ws, size_t ws_size,
                              hipStream_t stream)
{
    (void)in_sizes; (void)n_in; (void)out_size; (void)ws_size;
    const float* x    = (const float*)d_in[0];
    const float* ln_g = (const float*)d_in[1];
    const float* ln_b = (const float*)d_in[2];
    const float* W_h  = (const float*)d_in[3];
    const float* b_h  = (const float*)d_in[4];
    const float* W_qk = (const float*)d_in[5];
    const float* b_qk = (const float*)d_in[6];
    const float* os_g = (const float*)d_in[7];
    const float* os_b = (const float*)d_in[8];
    const float* rel  = (const float*)d_in[9];
    const float* W_o  = (const float*)d_in[10];
    const float* b_o  = (const float*)d_in[11];
    float* out = (float*)d_out;

    constexpr long long MR = 16384; // 4 * 4096 rows
    char* p = (char*)d_ws;
    auto alloc = [&](long long bytes) -> void* {
        void* r = (void*)p; p += (bytes + 255) & ~255ll; return r;
    };
    h16*   normed16 = (h16*)alloc(MR * 1024 * 2);
    h16*   wh16     = (h16*)alloc(1024ll * 4096 * 2);
    h16*   wqk16    = (h16*)alloc(1024ll * 128 * 2);
    h16*   wo16     = (h16*)alloc(2048ll * 1024 * 2);
    h16*   hidden16 = (h16*)alloc(MR * 4096 * 2);   // [v | gate]
    float* qkf      = (float*)alloc(MR * 128 * 4);
    h16*   qq       = (h16*)alloc(MR * 128 * 2);
    h16*   lq       = (h16*)alloc(MR * 128 * 2);
    h16*   qkh      = (h16*)alloc(MR * 128 * 2);
    h16*   lk       = (h16*)alloc(MR * 128 * 2);
    float* btab     = (float*)alloc(512 * 4);
    h16*   attn16   = (h16*)alloc(64ll * 256 * 256 * 2);
    h16*   q16      = (h16*)alloc(MR * 2048 * 2);   // quad_out, later gated out
    h16*   linkv16  = (h16*)alloc(4ll * 128 * 2048 * 2);
    h16*   l16      = (h16*)alloc(MR * 2048 * 2);   // lin_out

    // 1) weight casts + layernorm + bias table
    k_cast_f32_h16<<<(1024 * 4096 + 255) / 256, 256, 0, stream>>>(W_h, wh16, 1024ll * 4096);
    k_cast_f32_h16<<<(1024 * 128 + 255) / 256, 256, 0, stream>>>(W_qk, wqk16, 1024ll * 128);
    k_cast_f32_h16<<<(2048 * 1024 + 255) / 256, 256, 0, stream>>>(W_o, wo16, 2048ll * 1024);
    k_layernorm<<<16384, 256, 0, stream>>>(x, ln_g, ln_b, normed16, 1024);
    k_relbias<<<1, 512, 0, stream>>>(rel, btab);

    // 2) hidden = silu(normed @ W_h + b_h)   [16384 x 4096]
    gemm_wmma<false, false><<<dim3(32, 128, 1), 256, 0, stream>>>(
        normed16, wh16, 16384, 4096, 1024, 1024, 4096, 0ll, 0ll,
        EpiSiluH16{hidden16, b_h, 4096});

    // 3) qk = silu(normed @ W_qk + b_qk)     [16384 x 128]
    gemm_wmma<false, false><<<dim3(1, 128, 1), 256, 0, stream>>>(
        normed16, wqk16, 16384, 128, 1024, 1024, 128, 0ll, 0ll,
        EpiSiluF32{qkf, b_qk, 128});

    // 4) per-head affine
    k_heads<<<(int)(MR * 128 / 256), 256, 0, stream>>>(qkf, os_g, os_b, qq, lq, qkh, lk);

    // 5) attn = relu(qq @ qk^T / G + bias)^2 per (b,g)  [64 x 256 x 256]
    gemm_wmma<false, true><<<dim3(2, 2, 64), 256, 0, stream>>>(
        qq, qkh, 256, 256, 128, 128, 128, 256ll * 128, 256ll * 128,
        EpiAttn{attn16, btab});

    // 6) quad_out = attn @ v per group       [64 x 256 x 2048]
    gemm_wmma<false, false><<<dim3(16, 2, 64), 256, 0, stream>>>(
        attn16, hidden16, 256, 2048, 256, 256, 4096, 65536ll, 256ll * 4096,
        EpiScaleH16{q16, 1.0f, 2048, 256});

    // 7) lin_kv = lin_k^T @ v / n per batch  [4 x 128 x 2048]
    gemm_wmma<true, false><<<dim3(16, 1, 4), 256, 0, stream>>>(
        lk, hidden16, 128, 2048, 4096, 128, 4096, 4096ll * 128, 4096ll * 4096,
        EpiScaleH16{linkv16, 1.0f / 4096.0f, 2048, 128});

    // 8) lin_out = lin_q @ lin_kv per batch  [4 x 4096 x 2048]
    gemm_wmma<false, false><<<dim3(16, 32, 4), 256, 0, stream>>>(
        lq, linkv16, 4096, 2048, 128, 128, 2048, 4096ll * 128, 128ll * 2048,
        EpiScaleH16{l16, 1.0f, 2048, 4096});

    // 9) gated combine (in place into q16)
    k_combine<<<(int)(MR * 2048 / 256), 256, 0, stream>>>(hidden16, l16, q16);

    // 10) out = gated @ W_o + b_o + x        [16384 x 1024] f32
    gemm_wmma<false, false><<<dim3(8, 128, 1), 256, 0, stream>>>(
        q16, wo16, 16384, 1024, 2048, 2048, 1024, 0ll, 0ll,
        EpiFinal{out, b_o, x, 1024});
}
